// MultiHeadAttention_48722109005844
// MI455X (gfx1250) — matrix operand order repaired and checked by reading, not compiled, not run on hardware
//
#include <hip/hip_runtime.h>
#include <stdint.h>

#define S_LEN   2048
#define DM      1024
#define N_HEADS 16
#define DEPTH   64
#define BATCH   2
#define M_ROWS  (BATCH * S_LEN)   // 4096

typedef __attribute__((ext_vector_type(16))) __bf16   v16bf;
typedef __attribute__((ext_vector_type(8)))  float    v8f;
typedef __attribute__((ext_vector_type(4)))  unsigned u32x4;

union Frag {
  v16bf v;
  u32x4 q[2];
};

__device__ __forceinline__ uint16_t f2bf(float x) {
  union { float f; uint32_t u; } a; a.f = x;
  uint32_t r = a.u + 0x7fffu + ((a.u >> 16) & 1u);   // round-to-nearest-even
  return (uint16_t)(r >> 16);
}

// A fragment: 16x32 bf16 tile, row-major source (ld in elements).
// A layout (matches B): lane half h, element i -> k = i + 16h.
// Both halves are 16B-contiguous -> two b128 loads.
__device__ __forceinline__ v16bf load_frag_a(const uint16_t* base, int ld, int m0, int k0, int lane) {
  const int r = lane & 15, h = lane >> 4;
  const uint16_t* p = base + (size_t)(m0 + r) * ld + k0;
  Frag f;
  f.q[0] = *reinterpret_cast<const u32x4*>(p + 16 * h);
  f.q[1] = *reinterpret_cast<const u32x4*>(p + 16 * h + 8);
  return f.v;
}

// B fragment: 32x16 bf16 tile from K-contiguous storage src[n][k] (ld in elements).
// B layout (matches A, contiguous k order): lane half h, element i -> k = i + 16h; one 32B-contiguous run.
__device__ __forceinline__ v16bf load_frag_b(const uint16_t* baseT, int ld, int n0, int k0, int lane) {
  const int n = lane & 15, h = lane >> 4;
  const uint16_t* p = baseT + (size_t)(n0 + n) * ld + k0 + 16 * h;
  Frag f;
  f.q[0] = *reinterpret_cast<const u32x4*>(p);
  f.q[1] = *reinterpret_cast<const u32x4*>(p + 8);
  return f.v;
}

__device__ __forceinline__ v8f wmma_bf16(v16bf a, v16bf b, v8f c) {
  return __builtin_amdgcn_wmma_f32_16x16x32_bf16(false, a, false, b, (short)0, c, false, false);
}

// ---------------------------------------------------------------- converts
__global__ void cvt_f32_bf16(const float* __restrict__ in, uint16_t* __restrict__ out, int n) {
  int i = blockIdx.x * blockDim.x + threadIdx.x;
  if (i < n) out[i] = f2bf(in[i]);
}

// W (1024x1024 row-major, [k][n]) -> WT[n][k] bf16 (K-contiguous for B fragments)
__global__ void cvt_transpose_bf16(const float* __restrict__ W, uint16_t* __restrict__ WT) {
  int i = blockIdx.x * blockDim.x + threadIdx.x;
  int n = i >> 10, k = i & (DM - 1);
  if (i < DM * DM)
    WT[(size_t)n * DM + k] = f2bf(W[(size_t)k * DM + n]);
}

// ---------------------------------------------------------------- projection GEMM
// C(4096x1024) = A(4096x1024,bf16) * B(1024x1024 via BT bf16) + bias
// 8 waves: 4 along M x 2 along N; wave tile 32x64 -> 8 WMMA per 12 b128 loads per k-step.
// MODE 0: write bf16 in (B,H,S,d) layout (Q/K)
// MODE 2: write bf16 in (B,H,d,S) layout (V transposed)
// MODE 3: write fp32 row-major (final output)
template <int MODE>
__global__ void proj_gemm(const uint16_t* __restrict__ A, const uint16_t* __restrict__ BT,
                          const float* __restrict__ bias, void* __restrict__ outp) {
  const int lane = threadIdx.x & 31;
  const int w    = threadIdx.x >> 5;
  const int mw   = blockIdx.y * 128 + (w >> 1) * 32;   // 4 waves along M
  const int nw   = blockIdx.x * 128 + (w & 1) * 64;    // 2 waves along N

  v8f acc[2][4] = {};
  for (int k0 = 0; k0 < DM; k0 += 32) {
    v16bf a0 = load_frag_a(A, DM, mw,      k0, lane);
    v16bf a1 = load_frag_a(A, DM, mw + 16, k0, lane);
    v16bf bf[4];
    #pragma unroll
    for (int j = 0; j < 4; ++j) bf[j] = load_frag_b(BT, DM, nw + 16 * j, k0, lane);
    #pragma unroll
    for (int j = 0; j < 4; ++j) {
      acc[0][j] = wmma_bf16(a0, bf[j], acc[0][j]);
      acc[1][j] = wmma_bf16(a1, bf[j], acc[1][j]);
    }
  }

  const int h = lane >> 4, cn = lane & 15;
  #pragma unroll
  for (int j = 0; j < 4; ++j) {
    const int col = nw + j * 16 + cn;
    const float bv = bias[col];
    #pragma unroll
    for (int i = 0; i < 2; ++i) {
      #pragma unroll
      for (int r = 0; r < 8; ++r) {
        const int row = mw + i * 16 + r + 8 * h;   // C/D layout: VGPR r -> M = r (+8 for hi half)
        const float val = acc[i][j][r] + bv;
        if (MODE == 3) {
          ((float*)outp)[(size_t)row * DM + col] = val;
        } else {
          const int bb = row >> 11, s = row & (S_LEN - 1);
          const int hd = col >> 6,  d = col & (DEPTH - 1);
          const uint16_t bw = f2bf(val);
          if (MODE == 2)
            ((uint16_t*)outp)[(((size_t)(bb * N_HEADS + hd)) * DEPTH + d) * S_LEN + s] = bw;
          else
            ((uint16_t*)outp)[(((size_t)(bb * N_HEADS + hd)) * S_LEN + s) * DEPTH + d] = bw;
        }
      }
    }
  }
}

// ---------------------------------------------------------------- fused attention
// One workgroup = (b, h, 16 query rows). Logit row-block lives in LDS (200 KB of 320 KB).
__global__ void attn_fused(const uint16_t* __restrict__ Qh, const uint16_t* __restrict__ Kh,
                           const uint16_t* __restrict__ Vt, const float* __restrict__ mask,
                           float* __restrict__ attn_out, uint16_t* __restrict__ ctx_out) {
  extern __shared__ float smemf[];
  float*    slog  = smemf;                                        // 16*2048 f32 (128 KB)
  uint16_t* sattn = (uint16_t*)((char*)smemf + 16 * S_LEN * 4);   // 16*2048 bf16 (64 KB)
  float*    sctx  = (float*)((char*)sattn + 16 * S_LEN * 2);      // 16*64 f32 (4 KB)

  const int lane = threadIdx.x & 31;
  const int w    = threadIdx.x >> 5;
  const int q0   = blockIdx.x * 16;
  const int hh   = blockIdx.y;
  const int b    = blockIdx.z;
  const int h    = lane >> 4, cn = lane & 15;

  for (int i = threadIdx.x; i < 16 * DEPTH; i += 256) sctx[i] = 0.f;

  const uint16_t* Qp = Qh + (size_t)(b * N_HEADS + hh) * S_LEN * DEPTH;  // (S x d)
  const uint16_t* Kp = Kh + (size_t)(b * N_HEADS + hh) * S_LEN * DEPTH;  // (S x d) = B for QK^T
  const uint16_t* Vp = Vt + (size_t)(b * N_HEADS + hh) * DEPTH * S_LEN;  // (d x S) = B for attn*V
  const float*    mk = mask + (size_t)b * S_LEN;

  // ---- Stage A: logits = Q K^T / 8 + mask, each wave owns a 256-key strip
  v16bf a0 = load_frag_a(Qp, DEPTH, q0, 0,  lane);
  v16bf a1 = load_frag_a(Qp, DEPTH, q0, 32, lane);
  const int t0 = w * 256;
  for (int nt = 0; nt < 16; ++nt) {
    const int n0 = t0 + nt * 16;
    v16bf b0 = load_frag_b(Kp, DEPTH, n0, 0,  lane);
    v16bf b1 = load_frag_b(Kp, DEPTH, n0, 32, lane);
    v8f c = {};
    c = wmma_bf16(a0, b0, c);
    c = wmma_bf16(a1, b1, c);
    const float mval = mk[n0 + cn] * (-1e9f);
    #pragma unroll
    for (int r = 0; r < 8; ++r)
      slog[(r + 8 * h) * S_LEN + n0 + cn] = c[r] * 0.125f + mval;
  }
  __syncthreads();

  // ---- Stage B: row softmax (2 rows per wave), wave32 shuffle reductions
  for (int rr = 0; rr < 2; ++rr) {
    const int row = w * 2 + rr;
    float* L = slog + row * S_LEN;
    float mx = -3.0e38f;
    for (int i = lane; i < S_LEN; i += 32) mx = fmaxf(mx, L[i]);
    for (int off = 16; off > 0; off >>= 1) mx = fmaxf(mx, __shfl_xor(mx, off, 32));
    float sum = 0.f;
    for (int i = lane; i < S_LEN; i += 32) { float e = __expf(L[i] - mx); L[i] = e; sum += e; }
    for (int off = 16; off > 0; off >>= 1) sum += __shfl_xor(sum, off, 32);
    const float inv = 1.f / sum;
    float* arow = attn_out + (((size_t)(b * N_HEADS + hh)) * S_LEN + q0 + row) * S_LEN;
    uint16_t* srow = sattn + row * S_LEN;
    for (int i = lane; i < S_LEN; i += 32) {
      const float p = L[i] * inv;
      arow[i] = p;            // fp32 attn output (reference returns it)
      srow[i] = f2bf(p);      // bf16 operand for attn @ V
    }
  }
  __syncthreads();

  // ---- Stage C: ctx = attn @ V, K-dim (2048) split across the 8 waves
  v8f cacc[4] = {};
  for (int kk = 0; kk < 8; ++kk) {
    const int k0 = w * 256 + kk * 32;
    v16bf af = load_frag_a(sattn, S_LEN, 0, k0, lane);       // LDS b128 reads
    #pragma unroll
    for (int j = 0; j < 4; ++j) {
      v16bf bf = load_frag_b(Vp, S_LEN, j * 16, k0, lane);   // V^T rows are K-contiguous
      cacc[j] = wmma_bf16(af, bf, cacc[j]);
    }
  }
  #pragma unroll
  for (int j = 0; j < 4; ++j)
    #pragma unroll
    for (int r = 0; r < 8; ++r)
      atomicAdd(&sctx[(r + 8 * h) * DEPTH + j * 16 + cn], cacc[j][r]);   // ds_add_f32 reduce
  __syncthreads();

  for (int i = threadIdx.x; i < 16 * DEPTH; i += 256) {
    const int row = i >> 6, d = i & (DEPTH - 1);
    ctx_out[((size_t)(b * S_LEN) + q0 + row) * DM + hh * DEPTH + d] = f2bf(sctx[i]);
  }
}

// ---------------------------------------------------------------- host
extern "C" void kernel_launch(void* const* d_in, const int* in_sizes, int n_in,
                              void* d_out, int out_size, void* d_ws, size_t ws_size,
                              hipStream_t stream) {
  (void)in_sizes; (void)n_in; (void)out_size; (void)ws_size;
  const float* q    = (const float*)d_in[0];
  const float* k    = (const float*)d_in[1];
  const float* v    = (const float*)d_in[2];
  const float* mask = (const float*)d_in[3];
  const float* Wq   = (const float*)d_in[4];
  const float* bq   = (const float*)d_in[5];
  const float* Wk   = (const float*)d_in[6];
  const float* bk   = (const float*)d_in[7];
  const float* Wv   = (const float*)d_in[8];
  const float* bv   = (const float*)d_in[9];
  const float* Wo   = (const float*)d_in[10];
  const float* bo   = (const float*)d_in[11];

  char* ws = (char*)d_ws;
  const size_t MB = 1024 * 1024;
  uint16_t* qb   = (uint16_t*)(ws + 0 * MB);   // 8 MB each
  uint16_t* kb   = (uint16_t*)(ws + 8 * MB);
  uint16_t* vb   = (uint16_t*)(ws + 16 * MB);
  uint16_t* WqT  = (uint16_t*)(ws + 24 * MB);  // 2 MB each
  uint16_t* WkT  = (uint16_t*)(ws + 26 * MB);
  uint16_t* WvT  = (uint16_t*)(ws + 28 * MB);
  uint16_t* WoT  = (uint16_t*)(ws + 30 * MB);
  uint16_t* Qh   = (uint16_t*)(ws + 32 * MB);  // 8 MB each
  uint16_t* Kh   = (uint16_t*)(ws + 40 * MB);
  uint16_t* Vt   = (uint16_t*)(ws + 48 * MB);
  uint16_t* ctxb = (uint16_t*)(ws + 56 * MB);  // total 64 MB workspace

  float* outp  = (float*)d_out;
  float* attnp = outp + (size_t)M_ROWS * DM;

  const int nElem = M_ROWS * DM;   // 4 Mi
  cvt_f32_bf16<<<(nElem + 255) / 256, 256, 0, stream>>>(q, qb, nElem);
  cvt_f32_bf16<<<(nElem + 255) / 256, 256, 0, stream>>>(k, kb, nElem);
  cvt_f32_bf16<<<(nElem + 255) / 256, 256, 0, stream>>>(v, vb, nElem);
  const int nW = DM * DM;
  cvt_transpose_bf16<<<(nW + 255) / 256, 256, 0, stream>>>(Wq, WqT);
  cvt_transpose_bf16<<<(nW + 255) / 256, 256, 0, stream>>>(Wk, WkT);
  cvt_transpose_bf16<<<(nW + 255) / 256, 256, 0, stream>>>(Wv, WvT);
  cvt_transpose_bf16<<<(nW + 255) / 256, 256, 0, stream>>>(Wo, WoT);

  dim3 pg(DM / 128, M_ROWS / 128);   // (8, 32)
  proj_gemm<0><<<pg, 256, 0, stream>>>(qb, WqT, bq, Qh);
  proj_gemm<0><<<pg, 256, 0, stream>>>(kb, WkT, bk, Kh);
  proj_gemm<2><<<pg, 256, 0, stream>>>(vb, WvT, bv, Vt);

  const size_t smem = (size_t)16 * S_LEN * 4 + (size_t)16 * S_LEN * 2 + (size_t)16 * DEPTH * 4; // 200704
  hipFuncSetAttribute((const void*)attn_fused, hipFuncAttributeMaxDynamicSharedMemorySize, (int)smem);
  attn_fused<<<dim3(S_LEN / 16, N_HEADS, BATCH), 256, smem, stream>>>(Qh, Kh, Vt, mask, attnp, ctxb);

  proj_gemm<3><<<pg, 256, 0, stream>>>(ctxb, WoT, bo, (void*)outp);
}
